// EinFFT_6897717477703
// MI455X (gfx1250) — matrix operand order repaired and checked by reading, not compiled, not run on hardware
//
#include <hip/hip_runtime.h>
#include <hip/hip_bf16.h>
#include <math.h>

typedef _Float16 v16h __attribute__((ext_vector_type(16)));
typedef float    v8f  __attribute__((ext_vector_type(8)));
typedef int      gv4i __attribute__((vector_size(16)));   // GCC-vector int4 (builtin param type)

#ifndef __has_builtin
#define __has_builtin(x) 0
#endif

// gfx1250 async global->LDS copies (ASYNCcnt) with safe fallback.
#if defined(__AMDGCN__) && \
    __has_builtin(__builtin_amdgcn_global_load_async_to_lds_b128) && \
    __has_builtin(__builtin_amdgcn_s_wait_asynccnt)
#define USE_ASYNC_LDS 1
#else
#define USE_ASYNC_LDS 0
#endif

#if USE_ASYNC_LDS
__device__ __forceinline__ void async_cp16(float* lds_dst, const float* g_src) {
  __builtin_amdgcn_global_load_async_to_lds_b128(
      (__attribute__((address_space(1))) gv4i*)(void*)(g_src),
      (__attribute__((address_space(3))) gv4i*)(void*)(lds_dst), 0, 0);
}
#endif

// ---------------------------------------------------------------------------
// WMMA GEMM:  C[M,N] = A[M,K] @ W[N,K]^T   (fp32 in, f16 WMMA, fp32 acc/out)
// Block = 256 threads (8 waves as 4(M) x 2(N)). Block tile 64(M) x 64(N).
// Each wave: one A fragment reused across two 16x16 accumulators.
// Tiles staged fp32 in LDS (async DMA when available), cvt to f16 at frag build.
// ---------------------------------------------------------------------------
__launch_bounds__(256)
__global__ void gemm_f16_wmma(const float* __restrict__ A, const float* __restrict__ W,
                              float* __restrict__ C, int M, int N, int K) {
  __shared__ float Af[64 * 36];   // 64 rows x 32 k, stride 36 (16B-aligned, bank-spread)
  __shared__ float Wf[64 * 36];
  const int tid  = threadIdx.x;
  const int lane = tid & 31;
  const int wv   = tid >> 5;
  const int wvm  = wv & 3;        // 0..3  -> M sub-tile
  const int wvn  = wv >> 2;       // 0..1  -> N sub-strip (32 wide)
  const int m0   = blockIdx.y * 64;
  const int n0   = blockIdx.x * 64;

  // staging chunks: 512 x (4 floats); thread owns chunks tid and tid+256
  const int r0 = tid >> 3,         c0 = (tid & 7) * 4;
  const int r1 = (tid + 256) >> 3, c1 = (tid & 7) * 4;

  v8f acc0 = {0.f, 0.f, 0.f, 0.f, 0.f, 0.f, 0.f, 0.f};
  v8f acc1 = {0.f, 0.f, 0.f, 0.f, 0.f, 0.f, 0.f, 0.f};

  for (int k0 = 0; k0 < K; k0 += 32) {
    if (k0 + 32 < K) {  // prefetch next K tile (-> global_prefetch_b8)
      __builtin_prefetch(A + (size_t)(m0 + r0) * K + (k0 + 32) + c0, 0, 1);
      __builtin_prefetch(W + (size_t)(n0 + r0) * K + (k0 + 32) + c0, 0, 1);
    }
#if USE_ASYNC_LDS
    async_cp16(&Af[r0 * 36 + c0], A + (size_t)(m0 + r0) * K + k0 + c0);
    async_cp16(&Af[r1 * 36 + c1], A + (size_t)(m0 + r1) * K + k0 + c1);
    async_cp16(&Wf[r0 * 36 + c0], W + (size_t)(n0 + r0) * K + k0 + c0);
    async_cp16(&Wf[r1 * 36 + c1], W + (size_t)(n0 + r1) * K + k0 + c1);
    __builtin_amdgcn_s_wait_asynccnt(0);
#else
    *(float4*)&Af[r0 * 36 + c0] = *(const float4*)(A + (size_t)(m0 + r0) * K + k0 + c0);
    *(float4*)&Af[r1 * 36 + c1] = *(const float4*)(A + (size_t)(m0 + r1) * K + k0 + c1);
    *(float4*)&Wf[r0 * 36 + c0] = *(const float4*)(W + (size_t)(n0 + r0) * K + k0 + c0);
    *(float4*)&Wf[r1 * 36 + c1] = *(const float4*)(W + (size_t)(n0 + r1) * K + k0 + c1);
#endif
    __syncthreads();

    // Fragments per ISA 7.12.2 (16-bit A 16x32 / B 32x16, wave32)
    v16h a, b0, b1;
    {
      const int m  = wvm * 16 + (lane & 15);
      const int kb = (lane >> 4) * 8;        // lanes 0-15: K base 0; 16-31: base 8
      const float* pa = Af + m * 36;
#pragma unroll
      for (int e = 0; e < 8; ++e)  a[e] = (_Float16)pa[kb + e];        // K=kb..kb+7
#pragma unroll
      for (int e = 8; e < 16; ++e) a[e] = (_Float16)pa[kb + 8 + e];    // K=kb+16..kb+23
      const int ko = (lane >> 4) * 8;        // lanes 0-15: K 0..7,16..23; 16-31: 8..15,24..31
      const float* pb0 = Wf + (wvn * 32 + (lane & 15)) * 36 + ko;
      const float* pb1 = pb0 + 16 * 36;
#pragma unroll
      for (int e = 0; e < 16; ++e) b0[e] = (_Float16)pb0[e < 8 ? e : e + 8];
#pragma unroll
      for (int e = 0; e < 16; ++e) b1[e] = (_Float16)pb1[e < 8 ? e : e + 8];
    }
    acc0 = __builtin_amdgcn_wmma_f32_16x16x32_f16(false, a, false, b0,
                                                  (short)0, acc0, false, false);
    acc1 = __builtin_amdgcn_wmma_f32_16x16x32_f16(false, a, false, b1,
                                                  (short)0, acc1, false, false);
    __syncthreads();
  }

  // C/D layout: VGPR i -> M = i (+8 for lanes>=16), N = lane&15
  const int ng = n0 + wvn * 32 + (lane & 15);
  const int mg = m0 + wvm * 16 + ((lane >> 4) * 8);
#pragma unroll
  for (int i = 0; i < 8; ++i) C[(size_t)(mg + i) * N + ng]      = acc0[i];
#pragma unroll
  for (int i = 0; i < 8; ++i) C[(size_t)(mg + i) * N + ng + 16] = acc1[i];
}

// ---------------------------------------------------------------------------
// LayerNorm over rows of 64 (optionally with residual add). 4 rows / block.
// ---------------------------------------------------------------------------
__launch_bounds__(256)
__global__ void ln4(const float* __restrict__ X, const float* __restrict__ R,
                    const float* __restrict__ g, const float* __restrict__ b,
                    float* __restrict__ out) {
  __shared__ float s1[256], s2[256];
  const int tid = threadIdx.x;
  const int grp = tid >> 6;
  const int j   = tid & 63;
  const size_t row = (size_t)blockIdx.x * 4 + grp;
  float v = X[row * 64 + j];
  if (R) v += R[row * 64 + j];
  s1[tid] = v;
  s2[tid] = v * v;
  __syncthreads();
  for (int s = 32; s > 0; s >>= 1) {
    if (j < s) { s1[tid] += s1[tid + s]; s2[tid] += s2[tid + s]; }
    __syncthreads();
  }
  const float mean = s1[grp * 64] * (1.f / 64.f);
  const float var  = s2[grp * 64] * (1.f / 64.f) - mean * mean;
  const float rs   = rsqrtf(var + 1e-5f);
  out[row * 64 + j] = (v - mean) * rs * g[j] + b[j];
}

// ---------------------------------------------------------------------------
// Depthwise causal conv1d (L=4, kernel=4) + SiLU on xp half of XZ -> U
// ---------------------------------------------------------------------------
__launch_bounds__(256)
__global__ void conv_silu(const float* __restrict__ XZ, const float* __restrict__ cw,
                          const float* __restrict__ cb, float* __restrict__ U) {
  const int idx = blockIdx.x * 256 + threadIdx.x;   // bn*128 + c
  const int bn = idx >> 7;
  const int c  = idx & 127;
  float wk[4], xp[4];
#pragma unroll
  for (int k = 0; k < 4; ++k) wk[k] = cw[c * 4 + k];
  const float bia = cb[c];
#pragma unroll
  for (int t = 0; t < 4; ++t) xp[t] = XZ[(size_t)(bn * 4 + t) * 256 + c];
#pragma unroll
  for (int t = 0; t < 4; ++t) {
    float acc = bia;
#pragma unroll
    for (int k = 0; k < 4; ++k) {
      const int ti = t - 3 + k;
      if (ti >= 0) acc += xp[ti] * wk[k];
    }
    U[(size_t)(bn * 4 + t) * 128 + c] = acc * (1.f / (1.f + __expf(-acc)));
  }
}

// ---------------------------------------------------------------------------
// Selective-scan: one block per BN row, one thread per channel d (128).
// Unrolled over L=4; B/C tiles staged in LDS. Epilogue folds +u*D and z-gate.
// ---------------------------------------------------------------------------
__launch_bounds__(128)
__global__ void ssm_scan(const float* __restrict__ U, const float* __restrict__ DBC,
                         const float* __restrict__ XZ, const float* __restrict__ A_log,
                         const float* __restrict__ dtb, const float* __restrict__ Dp,
                         float* __restrict__ YP) {
  __shared__ float Bs[4 * 64], Cs[4 * 64];
  const int bn = blockIdx.x;
  const int d  = threadIdx.x;
  for (int i = d; i < 256; i += 128) {
    const int t = i >> 6, s = i & 63;
    Bs[i] = DBC[(size_t)(bn * 4 + t) * 256 + 128 + s];
    Cs[i] = DBC[(size_t)(bn * 4 + t) * 256 + 192 + s];
  }
  __syncthreads();

  float delta[4], u[4], du[4], y[4] = {0.f, 0.f, 0.f, 0.f};
  const float db = dtb[d];
#pragma unroll
  for (int t = 0; t < 4; ++t) {
    const float dp = DBC[(size_t)(bn * 4 + t) * 256 + d] + db;   // composite dt proj
    delta[t] = (dp > 20.f) ? dp : log1pf(__expf(dp));            // softplus
    u[t] = U[(size_t)(bn * 4 + t) * 128 + d];
    du[t] = delta[t] * u[t];
  }
  for (int s = 0; s < 64; ++s) {
    const float As = -__expf(A_log[d * 64 + s]);
    float h = 0.f;
#pragma unroll
    for (int t = 0; t < 4; ++t) {
      h = __expf(delta[t] * As) * h + du[t] * Bs[t * 64 + s];
      y[t] += h * Cs[t * 64 + s];
    }
  }
  const float Dd = Dp[d];
#pragma unroll
  for (int t = 0; t < 4; ++t) {
    const float z = XZ[(size_t)(bn * 4 + t) * 256 + 128 + d];
    YP[(size_t)(bn * 4 + t) * 128 + d] =
        (y[t] + u[t] * Dd) * (z * (1.f / (1.f + __expf(-z))));
  }
}

// ---------------------------------------------------------------------------
// Composite weight: Wc[0:128]  = dt_proj_w @ x_proj_w[0:4]   (128x128)
//                   Wc[128:256]= x_proj_w[4:132]             (B then C rows)
// ---------------------------------------------------------------------------
__global__ void build_wc(const float* __restrict__ dtw, const float* __restrict__ xpw,
                         float* __restrict__ Wc) {
  const int idx = blockIdx.x * 256 + threadIdx.x;
  if (idx >= 256 * 128) return;
  const int n = idx >> 7, k = idx & 127;
  if (n < 128) {
    float s = 0.f;
#pragma unroll
    for (int j = 0; j < 4; ++j) s += dtw[n * 4 + j] * xpw[j * 128 + k];
    Wc[idx] = s;
  } else {
    Wc[idx] = xpw[(size_t)(4 + (n - 128)) * 128 + k];
  }
}

// ---------------------------------------------------------------------------
// Fused MLP: OUT = relu(ENC @ W1^T + b1) @ W2^T + b2, hidden (32768x1024)
// never materialized (saves ~256 MB HBM traffic). 128 threads / 16-row tile.
// ---------------------------------------------------------------------------
__launch_bounds__(128)
__global__ void mlp_fused(const float* __restrict__ ENC, const float* __restrict__ W1,
                          const float* __restrict__ B1, const float* __restrict__ W2,
                          const float* __restrict__ B2, float* __restrict__ OUT) {
  __shared__ _Float16 encA[16 * 72];   // 16 x 64, padded stride
  __shared__ _Float16 T[16 * 40];      // 16 x 32 hidden tile, padded stride
  const int tid  = threadIdx.x;
  const int lane = tid & 31;
  const int wv   = tid >> 5;
  const size_t m0 = (size_t)blockIdx.x * 16;

  {
    const int r  = tid >> 3;
    const int cb = (tid & 7) * 8;
    const float* src = ENC + (m0 + r) * 64 + cb;
    _Float16* dst = encA + r * 72 + cb;
#pragma unroll
    for (int i = 0; i < 8; ++i) dst[i] = (_Float16)src[i];
  }
  __syncthreads();

  v16h aENC[2];
  {
    const int m  = lane & 15;
    const int kb = (lane >> 4) * 8;
#pragma unroll
    for (int kk = 0; kk < 2; ++kk) {
      const _Float16* p = encA + m * 72 + kk * 32;
#pragma unroll
      for (int e = 0; e < 8; ++e)  aENC[kk][e] = p[kb + e];
#pragma unroll
      for (int e = 8; e < 16; ++e) aENC[kk][e] = p[kb + 8 + e];
    }
  }

  v8f acc = {0.f, 0.f, 0.f, 0.f, 0.f, 0.f, 0.f, 0.f};
  const int n_lane = lane & 15;
  const int ko     = (lane >> 4) * 8;

  for (int hc = 0; hc < 1024; hc += 32) {
    if (wv < 2) {   // waves 0,1: hidden tile cols [16*wv .. 16*wv+15]
      v8f tacc = {0.f, 0.f, 0.f, 0.f, 0.f, 0.f, 0.f, 0.f};
      const int hh = hc + wv * 16 + n_lane;
#pragma unroll
      for (int kk = 0; kk < 2; ++kk) {
        v16h b;
        const float* q = W1 + (size_t)hh * 64 + kk * 32 + ko;
#pragma unroll
        for (int e = 0; e < 16; ++e) b[e] = (_Float16)q[e < 8 ? e : e + 8];
        tacc = __builtin_amdgcn_wmma_f32_16x16x32_f16(false, aENC[kk], false, b,
                                                      (short)0, tacc, false, false);
      }
      const float bia = B1[hh];
      const int mb = (lane >> 4) * 8;
#pragma unroll
      for (int i = 0; i < 8; ++i) {
        float v = tacc[i] + bia;
        v = v > 0.f ? v : 0.f;                                 // ReLU
        T[(mb + i) * 40 + wv * 16 + n_lane] = (_Float16)v;
      }
    }
    __syncthreads();

    v16h aT, bW;   // all 4 waves: acc += T @ W2_chunk^T
    {
      const int m  = lane & 15;
      const int kb = (lane >> 4) * 8;
      const _Float16* p = T + m * 40;
#pragma unroll
      for (int e = 0; e < 8; ++e)  aT[e] = p[kb + e];
#pragma unroll
      for (int e = 8; e < 16; ++e) aT[e] = p[kb + 8 + e];
      const int n = wv * 16 + n_lane;
      const float* q = W2 + (size_t)n * 1024 + hc + ko;
#pragma unroll
      for (int e = 0; e < 16; ++e) bW[e] = (_Float16)q[e < 8 ? e : e + 8];
    }
    acc = __builtin_amdgcn_wmma_f32_16x16x32_f16(false, aT, false, bW,
                                                 (short)0, acc, false, false);
    __syncthreads();
  }

  const int n  = wv * 16 + n_lane;
  const float bb = B2[n];
  const int mb = (lane >> 4) * 8;
#pragma unroll
  for (int i = 0; i < 8; ++i) OUT[(m0 + mb + i) * 64 + n] = acc[i] + bb;
}

// ---------------------------------------------------------------------------
extern "C" void kernel_launch(void* const* d_in, const int* in_sizes, int n_in,
                              void* d_out, int out_size, void* d_ws, size_t ws_size,
                              hipStream_t stream) {
  const float* x        = (const float*)d_in[0];
  // d_in[1]=H, d_in[2]=W (unused)
  const float* norm_g   = (const float*)d_in[3];
  const float* norm_b   = (const float*)d_in[4];
  const float* in_proj  = (const float*)d_in[5];   // (256,64)
  const float* conv_w   = (const float*)d_in[6];   // (128,1,4)
  const float* conv_b   = (const float*)d_in[7];   // (128)
  const float* x_proj   = (const float*)d_in[8];   // (132,128)
  const float* dt_w     = (const float*)d_in[9];   // (128,4)
  const float* dt_b     = (const float*)d_in[10];  // (128)
  const float* A_log    = (const float*)d_in[11];  // (128,64)
  const float* Dp       = (const float*)d_in[12];  // (128)
  const float* out_proj = (const float*)d_in[13];  // (64,128)
  const float* n3g = (const float*)d_in[14];
  const float* n3b = (const float*)d_in[15];
  const float* n4g = (const float*)d_in[16];
  const float* n4b = (const float*)d_in[17];
  const float* w1  = (const float*)d_in[18];       // (1024,64)
  const float* b1  = (const float*)d_in[19];
  const float* w2  = (const float*)d_in[20];       // (64,1024)
  const float* b2  = (const float*)d_in[21];
  float* out = (float*)d_out;

  // Workspace layout (floats). Peak ~101 MB with aliasing.
  float* ws   = (float*)d_ws;
  float* XZ   = ws;                    // 32768*256 = 8388608
  float* U    = XZ + 8388608;          // 32768*128 = 4194304
  float* DBC  = U + 4194304;           // 32768*256 = 8388608
  float* YP   = DBC + 8388608;         // 32768*128 = 4194304
  float* Wc   = YP + 4194304;          // 256*128   = 32768
  float* Xln  = YP;                    // alias: dead before YP is written
  float* XM   = XZ;                    // alias: XZ dead after scan
  float* ENC1 = XZ + 2097152;
  float* MLPO = XZ + 4194304;

  // out0 = x (untouched input)
  (void)hipMemcpyAsync(out, x, (size_t)2097152 * sizeof(float),
                       hipMemcpyDeviceToDevice, stream);

  build_wc<<<128, 256, 0, stream>>>(dt_w, x_proj, Wc);
  ln4<<<8192, 256, 0, stream>>>(x, nullptr, norm_g, norm_b, Xln);
  gemm_f16_wmma<<<dim3(4, 512), 256, 0, stream>>>(Xln, in_proj, XZ, 32768, 256, 64);
  conv_silu<<<4096, 256, 0, stream>>>(XZ, conv_w, conv_b, U);
  gemm_f16_wmma<<<dim3(4, 512), 256, 0, stream>>>(U, Wc, DBC, 32768, 256, 128);
  ssm_scan<<<8192, 128, 0, stream>>>(U, DBC, XZ, A_log, dt_b, Dp, YP);
  gemm_f16_wmma<<<dim3(1, 512), 256, 0, stream>>>(YP, out_proj, XM, 32768, 64, 128);
  ln4<<<8192, 256, 0, stream>>>(x, XM, n3g, n3b, ENC1);
  mlp_fused<<<2048, 128, 0, stream>>>(ENC1, w1, b1, w2, b2, MLPO);
  ln4<<<8192, 256, 0, stream>>>(ENC1, MLPO, n4g, n4b, out + 2097152);
}
